// WindowAttention_54614804136412
// MI455X (gfx1250) — compile-verified, matrix operand order checked by reading, not run on hardware
//
#include <hip/hip_runtime.h>
#include <stdint.h>

#define DIM 96
#define HEADS 3
#define HD 32
#define WIN 7
#define RES0 56
#define NPIX (RES0*RES0)      // 3136
#define TOPK 64
#define BATCH 32
#define NW 64                 // 8x8 windows
#define WS2 49
#define BWN (BATCH*NW)        // 2048
#define KTOT 128
#define SCALE_Q 0.17677669529663687f

typedef __attribute__((ext_vector_type(16))) __bf16 bf16x16;
typedef __attribute__((ext_vector_type(8)))  float  floatx8;

union FragU { bf16x16 v; unsigned short us[16]; uint4 q[2]; };
union Acc8  { floatx8 v; float f[8]; };
union U4U   { uint4 q; unsigned short us[8]; };

__device__ __forceinline__ unsigned short f32_to_bf16(float x){
  union { float f; unsigned u; } v; v.f = x;
  unsigned r = v.u + 0x7FFFu + ((v.u >> 16) & 1u);
  return (unsigned short)(r >> 16);
}

// Fragment loader (A-pattern): lane&15 -> row, 16 bf16 walk columns with the
// ISA 8/8 split -> two contiguous 16B LDS reads per lane.
//  - A operand:  load from row-major M x K tile.
//  - B operand:  load from row-major N x K (i.e. B^T) tile -- identical pattern.
__device__ __forceinline__ bf16x16 load_frag(const unsigned short* s, int ld,
                                             int row0, int col0, int lane){
  int m = lane & 15, g = lane >> 4;
  const unsigned short* p = s + (row0 + m) * ld + col0 + g * 8;
  FragU u;
  u.q[0] = *(const uint4*)(p);
  u.q[1] = *(const uint4*)(p + 16);
  return u.v;
}

__device__ __forceinline__ floatx8 wmma_bf16(bf16x16 a, bf16x16 b, floatx8 c){
  return __builtin_amdgcn_wmma_f32_16x16x32_bf16(false, a, false, b, (short)0, c,
                                                 false, false);
}

// ---------------- Kernel 0: convert weights to bf16 (keep N-major layout) -----
__global__ void prep_weights(const float* __restrict__ qkv_w,
                             const float* __restrict__ proj_w,
                             unsigned short* __restrict__ wqkv_bf,   // [288][96]
                             unsigned short* __restrict__ wproj_bf){ // [96][96]
  int i = blockIdx.x * 256 + threadIdx.x;
  if (i < 288*96){
    wqkv_bf[i] = f32_to_bf16(qkv_w[i]);
  } else if (i < 288*96 + 96*96){
    int j = i - 288*96;
    wproj_bf[j] = f32_to_bf16(proj_w[j]);
  }
}

// ---------------- Kernel 1: zero the padded Q buffer --------------------------
__global__ void zero_q(uint4* __restrict__ q){
  int i = blockIdx.x * 256 + threadIdx.x;
  q[i] = make_uint4(0u, 0u, 0u, 0u);
}

// ---------------- Kernel 2: top-64 fine-level indices per window --------------
__global__ void topk_kernel(int* __restrict__ idx0){
  __shared__ unsigned char sel[NPIX];
  __shared__ unsigned int red[256];
  int tid = threadIdx.x;
  int w = blockIdx.x;
  int cqh = (w >> 3) * WIN + 3, cqw = (w & 7) * WIN + 3;
  for (int i = tid; i < NPIX; i += 256) sel[i] = 0;
  __syncthreads();
  for (int t = 0; t < TOPK; ++t){
    unsigned best = 0xFFFFFFFFu;
    for (int j = tid; j < NPIX; j += 256){
      if (!sel[j]){
        int jh = j / RES0, jw = j - jh*RES0;
        int dh = cqh - jh, dw = cqw - jw;
        unsigned d2 = (unsigned)(dh*dh + dw*dw);
        unsigned key = (d2 << 12) | (unsigned)j;  // stable: ties -> lower index
        best = (key < best) ? key : best;
      }
    }
    red[tid] = best;
    __syncthreads();
    for (int s = 128; s > 0; s >>= 1){
      if (tid < s) red[tid] = (red[tid+s] < red[tid]) ? red[tid+s] : red[tid];
      __syncthreads();
    }
    if (tid == 0){
      int j = (int)(red[0] & 4095u);
      idx0[w*TOPK + t] = j;
      sel[j] = 1;
    }
    __syncthreads();
  }
}

// ---------------- Kernel 3: relative-position-bias MLP ------------------------
__global__ void rpb_kernel(const int* __restrict__ idx0,
                           const float* __restrict__ w1a, const float* __restrict__ b1a,
                           const float* __restrict__ w2a, const float* __restrict__ b2a,
                           const float* __restrict__ w1b, const float* __restrict__ b1b,
                           const float* __restrict__ w2b, const float* __restrict__ b2b,
                           float* __restrict__ rpb){ // [64][3][49][128]
  int w = blockIdx.x / WS2, s = blockIdx.x - w*WS2;
  int t = threadIdx.x;                  // 0..127
  int wi = s / WIN - 3, wj = s % WIN - 3;
  int cqh = (w >> 3) * WIN + 3, cqw = (w & 7) * WIN + 3;
  float relh, relw;
  const float *w1, *b1, *w2, *b2;
  if (t < TOPK){
    int j = idx0[w*TOPK + t];
    relh = (float)(cqh - j / RES0);
    relw = (float)(cqw - j % RES0);
    w1 = w1a; b1 = b1a; w2 = w2a; b2 = b2a;
  } else {
    int tc = t - TOPK;
    relh = (float)(cqh - ((tc >> 3) * WIN + 3));
    relw = (float)(cqw - ((tc & 7) * WIN + 3));
    w1 = w1b; b1 = b1b; w2 = w2b; b2 = b2b;
  }
  float ch = (relh + (float)wi) * (1.0f / 56.0f);
  float cw = (relw + (float)wj) * (1.0f / 56.0f);
  float o0 = b2[0], o1 = b2[1], o2 = b2[2];
#pragma unroll
  for (int d = 0; d < 32; ++d){
    float hh = fmaxf(0.0f, w1[d*2] * ch + w1[d*2 + 1] * cw + b1[d]);
    o0 += w2[d]      * hh;
    o1 += w2[32 + d] * hh;
    o2 += w2[64 + d] * hh;
  }
  int base = (w*HEADS)*WS2*KTOT + s*KTOT + t;
  rpb[base]                = o0;
  rpb[base + WS2*KTOT]     = o1;
  rpb[base + 2*WS2*KTOT]   = o2;
}

// ---------------- Kernel 4: QKV GEMM for x0 (WMMA) ----------------------------
// M = 32 rows/block (32 | 3136 -> block never crosses an image), N=288, K=96.
__global__ void __launch_bounds__(256) qkv0_gemm(
    const float* __restrict__ x0, const unsigned short* __restrict__ wqkv,
    const float* __restrict__ qkv_b,
    unsigned short* __restrict__ q_bf,   // [2048][3][64][32]
    unsigned short* __restrict__ k_bf,   // [32][3136][3][32]
    unsigned short* __restrict__ v_bf){
  __shared__ __align__(16) unsigned short As[32*96];   // M x K
  __shared__ __align__(16) unsigned short Bs[288*96];  // N x K (B^T, natural)
  int tid = threadIdx.x, lane = tid & 31, wave = tid >> 5;
  int g = lane >> 4, nl = lane & 15;
  int blk = blockIdx.x;
  int img = blk / 98;                    // batch image (3136/32 = 98 blocks)
  int pix_base = (blk - img*98) * 32;    // first pixel of this block's rows
  int mbase = blk * 32;
  for (int i = tid; i < 32*96; i += 256){
    int r = i / 96, c = i - r*96;
    As[r*96 + c] = f32_to_bf16(x0[(mbase + r)*96 + c]);
  }
  for (int i = tid; i < (288*96)/2; i += 256)
    ((unsigned int*)Bs)[i] = ((const unsigned int*)wqkv)[i];
  __syncthreads();
  for (int t = wave; t < 36; t += 8){
    int mt = t / 18, nt = t - mt*18;
    floatx8 accv = {0,0,0,0,0,0,0,0};
#pragma unroll
    for (int ks = 0; ks < 3; ++ks)
      accv = wmma_bf16(load_frag(As, 96, mt*16, ks*32, lane),
                       load_frag(Bs, 96, nt*16, ks*32, lane), accv);
    Acc8 acc; acc.v = accv;
    int n = nt*16 + nl;
    float bias = qkv_b[n];
    int j = nt / 6;                      // 0=q, 1=k, 2=v (wave-uniform)
    int d = (nt - j*6)*16 + nl;          // channel within q/k/v
    int head = d >> 5, hd = d & 31;
    int mrow = pix_base + mt*16 + g*8;   // first pixel of this lane's rows
    if (j == 0){
#pragma unroll
      for (int e = 0; e < 8; ++e){
        int p = mrow + e;
        int ph = p / RES0, pw = p - ph*RES0;
        int win = (ph/WIN)*8 + (pw/WIN);
        int rowin = (ph%WIN)*WIN + (pw%WIN);
        q_bf[(((img*NW + win)*HEADS + head)*64 + rowin)*HD + hd] =
            f32_to_bf16((acc.f[e] + bias) * SCALE_Q);
      }
    } else {
      unsigned short* dst = (j == 1) ? k_bf : v_bf;
      int base = ((img*NPIX + mrow)*HEADS + head)*HD + hd;
#pragma unroll
      for (int e = 0; e < 8; ++e)
        dst[base + e*HEADS*HD] = f32_to_bf16(acc.f[e] + bias);
    }
  }
}

// ---------------- Kernel 5: K/V projection for x1 (small, scalar) -------------
__global__ void qkv1_kernel(const float* __restrict__ x1,
                            const float* __restrict__ qkv_w,
                            const float* __restrict__ qkv_b,
                            unsigned short* __restrict__ k1_bf,  // [32*64][96]
                            unsigned short* __restrict__ v1_bf){
  int gid = blockIdx.x * 256 + threadIdx.x;   // 2048*192 total
  int row = gid / 192, c = gid - row*192;
  const float* xr = x1 + row * 96;
  const float* wr = qkv_w + (96 + c) * 96;
  float acc = qkv_b[96 + c];
#pragma unroll 8
  for (int k = 0; k < 96; ++k) acc += xr[k] * wr[k];
  unsigned short* dst = (c < 96) ? k1_bf : v1_bf;
  dst[row*96 + (c % 96)] = f32_to_bf16(acc);
}

// ---------------- Kernel 6: fused attention + projection (WMMA) ---------------
__global__ void __launch_bounds__(128) attn_proj(
    const unsigned short* __restrict__ q_bf,
    const unsigned short* __restrict__ k_bf,  const unsigned short* __restrict__ v_bf,
    const unsigned short* __restrict__ k1_bf, const unsigned short* __restrict__ v1_bf,
    const int* __restrict__ idx0, const float* __restrict__ rpb,
    const unsigned short* __restrict__ wproj, const float* __restrict__ proj_b,
    float* __restrict__ out){
  __shared__ __align__(16) unsigned short Qs[64*40];    // Q rows (M x K)
  __shared__ __align__(16) unsigned short KVs[128*40];  // K rows; later V^T (32x136)
  __shared__ __align__(16) unsigned short Ps[64*136];   // P rows (M x K)
  __shared__ __align__(16) unsigned short Os[64*104];   // attention out rows
  __shared__ __align__(16) unsigned short Wp[96*96];    // proj W, N x K natural
  int tid = threadIdx.x, lane = tid & 31, wave = tid >> 5;
  int g = lane >> 4, nl = lane & 15;
  int bw = blockIdx.x, b = bw >> 6, w = bw & 63;
  int mt = wave;
  for (int i = tid; i < (96*96)/2; i += 128)
    ((unsigned int*)Wp)[i] = ((const unsigned int*)wproj)[i];
  for (int h = 0; h < HEADS; ++h){
    // stage Q (64x32 bf16) and gathered K rows (128x32 bf16)
    const uint4* qsrc = (const uint4*)(q_bf + (bw*HEADS + h)*64*HD);
    for (int i = tid; i < 256; i += 128){
      int r = i >> 2, c = i & 3;
      *(uint4*)(Qs + r*40 + c*8) = qsrc[i];
    }
    {
      int t = tid;  // 128 threads -> 128 key rows
      const unsigned short* src = (t < TOPK)
        ? k_bf  + ((b*NPIX + idx0[w*TOPK + t])*HEADS + h)*HD
        : k1_bf + ((b*NW + (t - TOPK))*HEADS + h)*HD;
#pragma unroll
      for (int c = 0; c < 4; ++c)
        *(uint4*)(KVs + t*40 + c*8) = ((const uint4*)src)[c];
    }
    __syncthreads();
    // S = (Q*scale) . K^T : B-frag of K^T == A-pattern load of row-major K
    bf16x16 aq = load_frag(Qs, 40, mt*16, 0, lane);
    Acc8 s[8];
#pragma unroll
    for (int nt = 0; nt < 8; ++nt){
      floatx8 z = {0,0,0,0,0,0,0,0};
      s[nt].v = wmma_bf16(aq, load_frag(KVs, 40, nt*16, 0, lane), z);
    }
    const float* rpbp = rpb + (w*HEADS + h)*WS2*KTOT;
#pragma unroll
    for (int e = 0; e < 8; ++e){
      int m = mt*16 + g*8 + e;
      if (m < WS2){
#pragma unroll
        for (int nt = 0; nt < 8; ++nt)
          s[nt].f[e] += rpbp[m*KTOT + nt*16 + nl];
      }
      float rm = -1e30f;
#pragma unroll
      for (int nt = 0; nt < 8; ++nt) rm = fmaxf(rm, s[nt].f[e]);
#pragma unroll
      for (int off = 8; off >= 1; off >>= 1)
        rm = fmaxf(rm, __shfl_xor(rm, off, 32));
      float sum = 0.0f;
#pragma unroll
      for (int nt = 0; nt < 8; ++nt){
        float ex = __expf(s[nt].f[e] - rm);
        s[nt].f[e] = ex; sum += ex;
      }
#pragma unroll
      for (int off = 8; off >= 1; off >>= 1)
        sum += __shfl_xor(sum, off, 32);
      float inv = (m < WS2) ? (1.0f / sum) : 0.0f;   // padded rows -> zero P
#pragma unroll
      for (int nt = 0; nt < 8; ++nt)
        Ps[m*136 + nt*16 + nl] = f32_to_bf16(s[nt].f[e] * inv);
    }
    __syncthreads();
    // stage gathered V *transposed* into KVs: Vt[d][t], d<32, stride 136.
    // B-frag of V (K=t x N=d) == A-pattern load of V^T (N=d x K=t) row-major.
    {
      int t = tid;
      const unsigned short* src = (t < TOPK)
        ? v_bf  + ((b*NPIX + idx0[w*TOPK + t])*HEADS + h)*HD
        : v1_bf + ((b*NW + (t - TOPK))*HEADS + h)*HD;
      U4U r[4];
#pragma unroll
      for (int c = 0; c < 4; ++c) r[c].q = ((const uint4*)src)[c];
#pragma unroll
      for (int c = 0; c < 4; ++c)
#pragma unroll
        for (int d = 0; d < 8; ++d)
          KVs[(c*8 + d)*136 + t] = r[c].us[d];
    }
    __syncthreads();
    // O = P . V  (M=16/wave, N=32, K=128)
    Acc8 o[2];
    o[0].v = (floatx8){0,0,0,0,0,0,0,0};
    o[1].v = (floatx8){0,0,0,0,0,0,0,0};
#pragma unroll
    for (int ks = 0; ks < 4; ++ks){
      bf16x16 ap = load_frag(Ps, 136, mt*16, ks*32, lane);
#pragma unroll
      for (int ntv = 0; ntv < 2; ++ntv)
        o[ntv].v = wmma_bf16(ap, load_frag(KVs, 136, ntv*16, ks*32, lane), o[ntv].v);
    }
#pragma unroll
    for (int ntv = 0; ntv < 2; ++ntv)
#pragma unroll
      for (int e = 0; e < 8; ++e){
        int m = mt*16 + g*8 + e;
        Os[m*104 + h*HD + ntv*16 + nl] = f32_to_bf16(o[ntv].f[e]);
      }
    __syncthreads();
  }
  // projection: (64x96) @ proj_w^T + bias; Wp is N x K natural -> A-pattern B-frag
#pragma unroll
  for (int ntp = 0; ntp < 6; ++ntp){
    floatx8 pcv = {0,0,0,0,0,0,0,0};
#pragma unroll
    for (int ks = 0; ks < 3; ++ks)
      pcv = wmma_bf16(load_frag(Os, 104, mt*16, ks*32, lane),
                      load_frag(Wp, 96, ntp*16, ks*32, lane), pcv);
    Acc8 pc; pc.v = pcv;
    int n = ntp*16 + nl;
    float bias = proj_b[n];
#pragma unroll
    for (int e = 0; e < 8; ++e){
      int m = mt*16 + g*8 + e;
      if (m < WS2)
        out[(bw*WS2 + m)*DIM + n] = pc.f[e] + bias;
    }
  }
}

// ---------------- workspace layout -------------------------------------------
static constexpr size_t alignup(size_t x){ return (x + 255) & ~(size_t)255; }
static constexpr size_t SZ_WQKV = 288*96*2;
static constexpr size_t SZ_WPROJ = 96*96*2;
static constexpr size_t SZ_IDX  = (size_t)NW*TOPK*4;
static constexpr size_t SZ_RPB  = (size_t)NW*HEADS*WS2*KTOT*4;
static constexpr size_t SZ_Q    = (size_t)BWN*HEADS*64*HD*2;
static constexpr size_t SZ_K    = (size_t)BATCH*NPIX*HEADS*HD*2;
static constexpr size_t SZ_K1   = (size_t)BATCH*NW*HEADS*HD*2;

static constexpr size_t OFF_WQKV  = 0;
static constexpr size_t OFF_WPROJ = alignup(OFF_WQKV + SZ_WQKV);
static constexpr size_t OFF_IDX   = alignup(OFF_WPROJ + SZ_WPROJ);
static constexpr size_t OFF_RPB   = alignup(OFF_IDX + SZ_IDX);
static constexpr size_t OFF_Q     = alignup(OFF_RPB + SZ_RPB);
static constexpr size_t OFF_K     = alignup(OFF_Q + SZ_Q);
static constexpr size_t OFF_V     = alignup(OFF_K + SZ_K);
static constexpr size_t OFF_K1    = alignup(OFF_V + SZ_K);
static constexpr size_t OFF_V1    = alignup(OFF_K1 + SZ_K1);

extern "C" void kernel_launch(void* const* d_in, const int* in_sizes, int n_in,
                              void* d_out, int out_size, void* d_ws, size_t ws_size,
                              hipStream_t stream){
  const float* x0     = (const float*)d_in[0];
  const float* x1     = (const float*)d_in[1];
  const float* qkv_w  = (const float*)d_in[2];
  const float* qkv_b  = (const float*)d_in[3];
  const float* proj_w = (const float*)d_in[4];
  const float* proj_b = (const float*)d_in[5];
  const float* r0w1 = (const float*)d_in[6];
  const float* r0b1 = (const float*)d_in[7];
  const float* r0w2 = (const float*)d_in[8];
  const float* r0b2 = (const float*)d_in[9];
  const float* r1w1 = (const float*)d_in[10];
  const float* r1b1 = (const float*)d_in[11];
  const float* r1w2 = (const float*)d_in[12];
  const float* r1b2 = (const float*)d_in[13];
  float* out = (float*)d_out;

  char* ws = (char*)d_ws;
  unsigned short* wqkv_bf  = (unsigned short*)(ws + OFF_WQKV);
  unsigned short* wproj_bf = (unsigned short*)(ws + OFF_WPROJ);
  int*            idx0     = (int*)(ws + OFF_IDX);
  float*          rpb      = (float*)(ws + OFF_RPB);
  unsigned short* q_bf     = (unsigned short*)(ws + OFF_Q);
  unsigned short* k_bf     = (unsigned short*)(ws + OFF_K);
  unsigned short* v_bf     = (unsigned short*)(ws + OFF_V);
  unsigned short* k1_bf    = (unsigned short*)(ws + OFF_K1);
  unsigned short* v1_bf    = (unsigned short*)(ws + OFF_V1);

  prep_weights<<<dim3((288*96 + 96*96 + 255)/256), dim3(256), 0, stream>>>(
      qkv_w, proj_w, wqkv_bf, wproj_bf);
  zero_q<<<dim3((unsigned)(SZ_Q/16/256)), dim3(256), 0, stream>>>((uint4*)q_bf);
  topk_kernel<<<dim3(NW), dim3(256), 0, stream>>>(idx0);
  rpb_kernel<<<dim3(NW*WS2), dim3(128), 0, stream>>>(
      idx0, r0w1, r0b1, r0w2, r0b2, r1w1, r1b1, r1w2, r1b2, rpb);
  qkv0_gemm<<<dim3((BATCH*NPIX)/32), dim3(256), 0, stream>>>(
      x0, wqkv_bf, qkv_b, q_bf, k_bf, v_bf);
  qkv1_kernel<<<dim3((BATCH*NW*192)/256), dim3(256), 0, stream>>>(
      x1, qkv_w, qkv_b, k1_bf, v1_bf);
  attn_proj<<<dim3(BWN), dim3(128), 0, stream>>>(
      q_bf, k_bf, v_bf, k1_bf, v1_bf, idx0, rpb, wproj_bf, proj_b, out);
}
